// GraphDecoder_43009802502763
// MI455X (gfx1250) — matrix operand order repaired and checked by reading, not compiled, not run on hardware
//
#include <hip/hip_runtime.h>

#define B_     256
#define LAT    256
#define NA     64
#define EMB_   64
#define NBOND_ 5
#define HW     1000.0f

typedef float    v2f  __attribute__((ext_vector_type(2)));
typedef float    v8f  __attribute__((ext_vector_type(8)));
typedef _Float16 h2   __attribute__((ext_vector_type(2)));
typedef _Float16 v16h __attribute__((ext_vector_type(16)));

// ---------------------------------------------------------------------------
// K1: h[256,4096] = z[256,256] @ W_nodes[256,4096] + b_nodes   (fp32 WMMA)
// One wave computes one 16x16 tile; 4096 tiles = 512 blocks * 8 waves.
// ---------------------------------------------------------------------------
__global__ __launch_bounds__(256) void k1_gemm_h(const float* __restrict__ z,
                                                 const float* __restrict__ W,
                                                 const float* __restrict__ bias,
                                                 float* __restrict__ h) {
    const int lane = threadIdx.x & 31;
    const int wv   = threadIdx.x >> 5;
    const int tile = blockIdx.x * 8 + wv;      // 16 x 256 tiles
    const int m0   = (tile >> 8) << 4;
    const int n0   = (tile & 255) << 4;
    const int half = lane >> 4;                // K half-select
    const int l16  = lane & 15;
    const int row  = m0 + l16;                 // A row (M)
    const int col  = n0 + l16;                 // B/D col (N)

    v8f acc = {};
    for (int k = 0; k < LAT; k += 4) {
        const int kk = k + 2 * half;
        v2f a; a.x = z[row * LAT + kk];        a.y = z[row * LAT + kk + 1];
        v2f b; b.x = W[kk * 4096 + col];       b.y = W[(kk + 1) * 4096 + col];
        acc = __builtin_amdgcn_wmma_f32_16x16x4_f32(false, a, false, b,
                                                    (short)0, acc, false, false);
    }
    const float bv = bias[col];
#pragma unroll
    for (int r = 0; r < 8; ++r) {
        const int m = m0 + r + 8 * half;       // D row layout: lanes>=16 -> M+8
        h[m * 4096 + col] = acc[r] + bv;
    }
}

// ---------------------------------------------------------------------------
// K2: [a|c][16384,128] = h[16384,64] @ [W1a | W1b]   (fp32 WMMA)
// 1024 x 8 tiles = 1024 blocks * 8 waves. Columns <64 -> a, >=64 -> c.
// ---------------------------------------------------------------------------
__global__ __launch_bounds__(256) void k2_gemm_ac(const float* __restrict__ h,
                                                  const float* __restrict__ W1,
                                                  float* __restrict__ abuf,
                                                  float* __restrict__ cbuf) {
    const int lane = threadIdx.x & 31;
    const int wv   = threadIdx.x >> 5;
    const int tile = blockIdx.x * 8 + wv;
    const int m0   = (tile >> 3) << 4;
    const int n0   = (tile & 7) << 4;          // 0..112
    const int half = lane >> 4;
    const int l16  = lane & 15;
    const int row  = m0 + l16;
    const int col  = n0 + l16;                 // 0..127
    // Wcat[e][col] : col<64 -> W1[e][col] ; col>=64 -> W1[64+e][col-64]
    const float* Wb = (col < 64) ? (W1 + col) : (W1 + 64 * EMB_ + (col - 64));

    v8f acc = {};
    for (int k = 0; k < EMB_; k += 4) {
        const int kk = k + 2 * half;
        v2f a; a.x = h[row * EMB_ + kk];  a.y = h[row * EMB_ + kk + 1];
        v2f b; b.x = Wb[kk * EMB_];       b.y = Wb[(kk + 1) * EMB_];
        acc = __builtin_amdgcn_wmma_f32_16x16x4_f32(false, a, false, b,
                                                    (short)0, acc, false, false);
    }
    float* dst   = (n0 < 64) ? abuf : cbuf;
    const int cc = (n0 < 64) ? col : (col - 64);
#pragma unroll
    for (int r = 0; r < 8; ++r)
        dst[(m0 + r + 8 * half) * EMB_ + cc] = acc[r];
}

// ---------------------------------------------------------------------------
// K3: one block per batch b. Fused relu(a_i + c_j + b1) @ W2 via f16 WMMA,
// then masked symmetric combine. LDS: a'(f16) 8K + c(f16) 8K + logits(f16) 40K.
// ---------------------------------------------------------------------------
__global__ __launch_bounds__(256) void k3_epilogue(const float* __restrict__ abuf,
                                                   const float* __restrict__ cbuf,
                                                   const float* __restrict__ b1v,
                                                   const float* __restrict__ W2,
                                                   const float* __restrict__ b2,
                                                   const int*   __restrict__ nreal,
                                                   float* __restrict__ out) {
    __shared__ h2       a_s[NA * EMB_ / 2];        // a[b] + b1, packed f16 pairs
    __shared__ h2       c_s[NA * EMB_ / 2];        // c[b]
    __shared__ _Float16 L[NA * NA * NBOND_];       // unsymmetrized logits

    const int b   = blockIdx.x;
    const int tid = threadIdx.x;

    // ---- phase 0: stage a'=a+b1 and c into LDS as f16 pairs ----
    for (int idx = tid; idx < NA * EMB_ / 2; idx += 256) {
        const int i = idx >> 5;
        const int f = (idx & 31) << 1;
        h2 av, cv;
        av.x = (_Float16)(abuf[b * 4096 + i * 64 + f]     + b1v[f]);
        av.y = (_Float16)(abuf[b * 4096 + i * 64 + f + 1] + b1v[f + 1]);
        cv.x = (_Float16)(cbuf[b * 4096 + i * 64 + f]);
        cv.y = (_Float16)(cbuf[b * 4096 + i * 64 + f + 1]);
        a_s[idx] = av;
        c_s[idx] = cv;
    }
    __syncthreads();

    const int lane = tid & 31;
    const int wv   = tid >> 5;
    const int half = lane >> 4;
    const int l16  = lane & 15;

    // ---- build B fragments: W2 (64x5) zero-padded to 64x16, two K=32 chunks
    // B in contiguous k order used here: lanes<16 col=lane K=2r,2r+1 ; lanes>=16 K=16+2r
    v16h B0, B1;
#pragma unroll
    for (int r = 0; r < 8; ++r) {
        const int kb = 2 * r + (half ? 16 : 0);
        _Float16 x0 = 0, x1 = 0, y0 = 0, y1 = 0;
        if (l16 < NBOND_) {
            x0 = (_Float16)W2[(kb)      * NBOND_ + l16];
            x1 = (_Float16)W2[(kb + 1)  * NBOND_ + l16];
            y0 = (_Float16)W2[(32 + kb)     * NBOND_ + l16];
            y1 = (_Float16)W2[(32 + kb + 1) * NBOND_ + l16];
        }
        B0[2 * r] = x0; B0[2 * r + 1] = x1;
        B1[2 * r] = y0; B1[2 * r + 1] = y1;
    }

    // ---- phase 1: 256 tasks (j, i-tile); 8 waves x 32 tasks ----
    for (int t = wv; t < 256; t += 8) {
        const int j  = t & 63;
        const int i0 = (t >> 6) << 4;
        const int i  = i0 + l16;               // A row (M = lane%16)

        // A in contiguous k order used here: pair index p = r + (lane<16?0:8)
        v16h A0, A1;
#pragma unroll
        for (int r = 0; r < 8; ++r) {
            const int p = r + (half ? 8 : 0);
            h2 s  = a_s[i * 32 + p]      + c_s[j * 32 + p];       // v_pk_add_f16
            h2 s2 = a_s[i * 32 + p + 16] + c_s[j * 32 + p + 16];
            s.x  = s.x  > (_Float16)0 ? s.x  : (_Float16)0;       // relu
            s.y  = s.y  > (_Float16)0 ? s.y  : (_Float16)0;
            s2.x = s2.x > (_Float16)0 ? s2.x : (_Float16)0;
            s2.y = s2.y > (_Float16)0 ? s2.y : (_Float16)0;
            A0[2 * r] = s.x;  A0[2 * r + 1] = s.y;
            A1[2 * r] = s2.x; A1[2 * r + 1] = s2.y;
        }
        v8f acc = {};
        acc = __builtin_amdgcn_wmma_f32_16x16x32_f16(false, A0, false, B0,
                                                     (short)0, acc, false, false);
        acc = __builtin_amdgcn_wmma_f32_16x16x32_f16(false, A1, false, B1,
                                                     (short)0, acc, false, false);
        if (l16 < NBOND_) {
#pragma unroll
            for (int r = 0; r < 8; ++r) {
                const int io = i0 + r + 8 * half;
                L[(io * 64 + j) * NBOND_ + l16] = (_Float16)acc[r];
            }
        }
    }
    __syncthreads();

    // ---- phase 2: mask + symmetrize + bias, write output ----
    const int nra = nreal[b];
    for (int e = tid; e < NA * NA * NBOND_; e += 256) {
        const int k   = e % NBOND_;
        const int rem = e / NBOND_;
        const int j   = rem & 63;
        const int i   = rem >> 6;
        const bool msk = (i == j) || (i >= nra) || (j >= nra);
        float v;
        if (msk) {
            v = (k == NBOND_ - 1) ? HW : -HW;
        } else {
            v = 0.5f * ((float)L[(i * 64 + j) * NBOND_ + k] +
                        (float)L[(j * 64 + i) * NBOND_ + k]) + b2[k];
        }
        out[((b * 64 + i) * 64 + j) * NBOND_ + k] = v;
    }
}

// ---------------------------------------------------------------------------
extern "C" void kernel_launch(void* const* d_in, const int* in_sizes, int n_in,
                              void* d_out, int out_size, void* d_ws, size_t ws_size,
                              hipStream_t stream) {
    const float* z   = (const float*)d_in[0];
    const int*   nra = (const int*)  d_in[1];
    const float* Wn  = (const float*)d_in[2];
    const float* bn  = (const float*)d_in[3];
    const float* W1  = (const float*)d_in[4];
    const float* b1  = (const float*)d_in[5];
    const float* W2  = (const float*)d_in[6];
    const float* b2  = (const float*)d_in[7];
    float* out = (float*)d_out;

    float* h    = (float*)d_ws;            // 256*4096 floats (4 MB)
    float* abuf = h + B_ * 4096;           // 256*64*64 floats (4 MB)
    float* cbuf = abuf + B_ * NA * EMB_;   // 4 MB

    k1_gemm_h  <<<512,  256, 0, stream>>>(z, Wn, bn, h);
    k2_gemm_ac <<<1024, 256, 0, stream>>>(h, W1, abuf, cbuf);
    k3_epilogue<<<256,  256, 0, stream>>>(abuf, cbuf, b1, W2, b2, nra, out);
}
